// SelfAttention_16879221473647
// MI455X (gfx1250) — compile-verified, matrix operand order checked by reading, not run on hardware
//
#include <hip/hip_runtime.h>

#define BATCH 8
#define SEQ   2048
#define DIM   512

typedef __bf16 bf16_t;
typedef __attribute__((ext_vector_type(16))) __bf16        v16bf;
typedef __attribute__((ext_vector_type(8)))  float         v8f;
typedef __attribute__((ext_vector_type(4)))  unsigned int  v4u;
typedef __attribute__((ext_vector_type(8)))  unsigned int  v8u;
typedef __attribute__((ext_vector_type(4)))  int           v4i;
typedef __attribute__((ext_vector_type(8)))  int           v8i;

// LDS row strides (elements) chosen so TDM padding kills bank conflicts
#define K_LDS_LD 520   // 32 keys x 512 dims + 16B pad per 1024B row  -> 1040B stride
#define V_LDS_LD 40    // 512 dims x 32 keys + 16B pad per 64B row    -> 80B stride

// ---------- WMMA fragment loaders ----------
// ISA 16-bit A(16x32) layout: lanes 0-15 hold rows M=lane with K={k0..k0+7, k0+16..k0+23},
// lanes 16-31 hold rows M=lane-16 with K={k0+8..k0+15, k0+24..k0+31}.
// B(32x16) mirrors with N on lanes; both reduce to the same per-lane addressing when the
// "major" dimension (M rows for A, N cols for B) is the leading dim of a row-major matrix.
static __device__ __forceinline__ v16bf load_frag_bf16(const bf16_t* mat, int major0,
                                                       int k0, int ld) {
  const int lane = threadIdx.x & 31;
  const int half = lane >> 4;
  const bf16_t* p = mat + (long)(major0 + (lane & 15)) * ld + k0 + half * 8;
  v4u a = *(const v4u*)p;
  v4u b = *(const v4u*)(p + 16);
  v8u u;
  u[0] = a[0]; u[1] = a[1]; u[2] = a[2]; u[3] = a[3];
  u[4] = b[0]; u[5] = b[1]; u[6] = b[2]; u[7] = b[3];
  return __builtin_bit_cast(v16bf, u);
}

// f32 source: load 16 floats, convert via native packed bf16 converts.
static __device__ __forceinline__ v16bf load_frag_f32(const float* mat, int major0,
                                                      int k0, int ld) {
  const int lane = threadIdx.x & 31;
  const int half = lane >> 4;
  const float* p = mat + (long)(major0 + (lane & 15)) * ld + k0 + half * 8;
  float4 a = *(const float4*)p;
  float4 b = *(const float4*)(p + 4);
  float4 c = *(const float4*)(p + 16);
  float4 d = *(const float4*)(p + 20);
  v16bf r;
  r[0]  = (__bf16)a.x; r[1]  = (__bf16)a.y; r[2]  = (__bf16)a.z; r[3]  = (__bf16)a.w;
  r[4]  = (__bf16)b.x; r[5]  = (__bf16)b.y; r[6]  = (__bf16)b.z; r[7]  = (__bf16)b.w;
  r[8]  = (__bf16)c.x; r[9]  = (__bf16)c.y; r[10] = (__bf16)c.z; r[11] = (__bf16)c.w;
  r[12] = (__bf16)d.x; r[13] = (__bf16)d.y; r[14] = (__bf16)d.z; r[15] = (__bf16)d.w;
  return r;
}

static __device__ __forceinline__ v8f wmma_bf16(v16bf a, v16bf b, v8f c) {
  return __builtin_amdgcn_wmma_f32_16x16x32_bf16(false, a, false, b, (short)0, c,
                                                 false, false);
}

// ---------- Tensor Data Mover: 2D bf16 tile, global -> LDS ----------
// Descriptor packing per CDNA5 ISA 8.3/8.4. data_size=1 (2 bytes/elem).
// pad_interval/pad_amount add LDS padding to avoid bank conflicts.
static __device__ __forceinline__ void tdm_load_2d_bf16(
    unsigned lds_addr, const bf16_t* gptr,
    unsigned tile_d0, unsigned tile_d1,
    unsigned long long stride0_elems,
    unsigned pad_interval_code, unsigned pad_amount_code) {
  const unsigned long long ga = (unsigned long long)(uintptr_t)gptr;
  v4u g0;
  g0[0] = 1u;                                   // count=1, user descriptor
  g0[1] = lds_addr;                             // lds_addr @ bits 63:32
  g0[2] = (unsigned)(ga & 0xFFFFFFFFu);         // global_addr[31:0]
  g0[3] = (unsigned)((ga >> 32) & 0x01FFFFFFu)  // global_addr[56:32]
          | 0x80000000u;                        // type=2 ("image") @ bits 127:126
  v8i g1;
  g1[0] = (int)((1u << 16)                      // data_size=1 -> 2 bytes
                | (1u << 20)                    // pad_enable
                | (pad_interval_code << 22)     // DWORDs between pads
                | (pad_amount_code << 25));     // pad DWORDs - 1
  g1[1] = (int)((tile_d0 & 0xFFFFu) << 16);     // tensor_dim0[15:0] (= tile width)
  g1[2] = (int)(((tile_d0 >> 16) & 0xFFFFu) | ((tile_d1 & 0xFFFFu) << 16));
  g1[3] = (int)(((tile_d1 >> 16) & 0xFFFFu) | ((tile_d0 & 0xFFFFu) << 16));  // tile_dim0
  g1[4] = (int)(tile_d1 & 0xFFFFu);             // tile_dim1 (tile_dim2 = 0)
  g1[5] = (int)(unsigned)(stride0_elems & 0xFFFFFFFFu);
  g1[6] = (int)(unsigned)((stride0_elems >> 32) & 0xFFFFu);
  g1[7] = 0;
  const v4i z4 = {0, 0, 0, 0};
  const v8i z8 = {0, 0, 0, 0, 0, 0, 0, 0};
  __builtin_amdgcn_tensor_load_to_lds(g0, g1, z4, z4, z8, 0);
}

static __device__ __forceinline__ unsigned lds_off(const void* p) {
  return (unsigned)(uintptr_t)p;  // low 32 bits of generic LDS address = LDS byte offset
}

// =====================================================================
// Phase 1: y = (x @ W^T + b) * scale,  f32 in -> bf16 out.
// =====================================================================
__global__ __launch_bounds__(128, 2) void proj_gemm_kernel(
    const float* __restrict__ x, const float* __restrict__ w,
    const float* __restrict__ bias, bf16_t* __restrict__ out,
    float scale, int transpose_out) {
  const int wave = threadIdx.x >> 5;
  const int lane = threadIdx.x & 31;
  const int half = lane >> 4;
  const int row0 = blockIdx.x * 16;
  const int col0 = (blockIdx.y * 4 + wave) * 16;

  v8f acc = {};
#pragma unroll
  for (int k0 = 0; k0 < DIM; k0 += 32) {
    v16bf af = load_frag_f32(x, row0, k0, DIM);
    v16bf bf = load_frag_f32(w, col0, k0, DIM);
    acc = wmma_bf16(af, bf, acc);
  }

  const int col = col0 + (lane & 15);
  const float bv = bias[col];
#pragma unroll
  for (int i = 0; i < 8; ++i) {
    const int row = row0 + i + half * 8;
    const bf16_t h = (bf16_t)((acc[i] + bv) * scale);
    if (transpose_out) {
      const int b = row >> 11;
      const int s = row & (SEQ - 1);
      out[((long)(b * DIM + col)) * SEQ + s] = h;
    } else {
      out[(long)row * DIM + col] = h;
    }
  }
}

// =====================================================================
// Phase 2: flash attention, TDM-fed. Block = 4 waves x 16 queries.
// Per 32-key chunk: TDM DMAs K[32x512] and V'[512x32] tiles into LDS
// (double-buffered, padded); all waves consume via ds_load_b128.
// =====================================================================
__global__ __launch_bounds__(128, 1) void attn_kernel(
    const bf16_t* __restrict__ qp,   // [B*SEQ, DIM] bf16, pre-scaled by 1/sqrt(D)
    const bf16_t* __restrict__ kp,   // [B*SEQ, DIM] bf16
    const bf16_t* __restrict__ vpT,  // [B, DIM, SEQ] bf16
    float* __restrict__ out) {       // [B, SEQ, DIM] f32
  __shared__ __align__(16) bf16_t Kbuf[2][32 * K_LDS_LD];   // 2 x 33280 B
  __shared__ __align__(16) bf16_t Vbuf[2][DIM * V_LDS_LD];  // 2 x 40960 B
  __shared__ __align__(16) bf16_t Pbuf[4][16 * 32];         // per-wave P staging

  const int wave = threadIdx.x >> 5;
  const int lane = threadIdx.x & 31;
  const int half = lane >> 4;
  const int b    = blockIdx.y;
  const int q0   = (blockIdx.x * 4 + wave) * 16;

  const bf16_t* qb = qp + ((long)b * SEQ) * DIM;
  const bf16_t* kb = kp + ((long)b * SEQ) * DIM;
  const bf16_t* vb = vpT + ((long)b * DIM) * SEQ;
  const bf16_t* Pl = &Pbuf[wave][0];

  // Hoist all Q fragments (16 rows x K=512 -> 16 fragments).
  v16bf qf[16];
#pragma unroll
  for (int t = 0; t < 16; ++t) qf[t] = load_frag_bf16(qb, q0, t * 32, DIM);

  const v8f vzero = {};
  v8f acc[32];
#pragma unroll
  for (int t = 0; t < 32; ++t) acc[t] = vzero;

  float mrow[8], lrow[8];
#pragma unroll
  for (int i = 0; i < 8; ++i) { mrow[i] = -3.0e38f; lrow[i] = 0.0f; }

  // Prologue: DMA chunk 0 into buffer 0 (wave 0 only; TDM ignores EXEC).
  if (wave == 0) {
    tdm_load_2d_bf16(lds_off(&Kbuf[0][0]), kb, DIM, 32, DIM, 7u, 3u);      // pad 16B/1024B
    tdm_load_2d_bf16(lds_off(&Vbuf[0][0]), vb, 32, DIM, SEQ, 3u, 3u);      // pad 16B/64B
  }

  for (int j = 0; j < SEQ; j += 32) {
    const int buf = (j >> 5) & 1;
    if (wave == 0) __builtin_amdgcn_s_wait_tensorcnt(0);  // chunk j landed
    __syncthreads();                                      // publish buf; prev reads done
    if (wave == 0 && j + 32 < SEQ) {                      // prefetch chunk j+1
      tdm_load_2d_bf16(lds_off(&Kbuf[buf ^ 1][0]), kb + (long)(j + 32) * DIM,
                       DIM, 32, DIM, 7u, 3u);
      tdm_load_2d_bf16(lds_off(&Vbuf[buf ^ 1][0]), vb + (j + 32),
                       32, DIM, SEQ, 3u, 3u);
    }
    const bf16_t* Kb = &Kbuf[buf][0];
    const bf16_t* Vb = &Vbuf[buf][0];

    // ---- scores: S[16q x 32k] over full 512 feature dim (from LDS) ----
    v8f s0 = vzero, s1 = vzero;
#pragma unroll
    for (int t = 0; t < 16; ++t) {
      v16bf k0f = load_frag_bf16(Kb, 0, t * 32, K_LDS_LD);   // keys j..j+15
      v16bf k1f = load_frag_bf16(Kb, 16, t * 32, K_LDS_LD);  // keys j+16..j+31
      s0 = wmma_bf16(qf[t], k0f, s0);
      s1 = wmma_bf16(qf[t], k1f, s1);
    }

    // ---- online softmax (row stats per (half, vgpr) = query row) ----
#pragma unroll
    for (int i = 0; i < 8; ++i) {
      float v = fmaxf(s0[i], s1[i]);
      v = fmaxf(v, __shfl_xor(v, 1, 32));
      v = fmaxf(v, __shfl_xor(v, 2, 32));
      v = fmaxf(v, __shfl_xor(v, 4, 32));
      v = fmaxf(v, __shfl_xor(v, 8, 32));
      const float mn   = fmaxf(mrow[i], v);
      const float corr = __expf(mrow[i] - mn);
      const float p0   = __expf(s0[i] - mn);
      const float p1   = __expf(s1[i] - mn);
      s0[i] = p0; s1[i] = p1;
      float rs = p0 + p1;
      rs += __shfl_xor(rs, 1, 32);
      rs += __shfl_xor(rs, 2, 32);
      rs += __shfl_xor(rs, 4, 32);
      rs += __shfl_xor(rs, 8, 32);
      lrow[i] = lrow[i] * corr + rs;
      mrow[i] = mn;
#pragma unroll
      for (int t = 0; t < 32; ++t) acc[t][i] *= corr;
    }

    // ---- re-layout P (D-fragment -> A-fragment) through LDS ----
#pragma unroll
    for (int i = 0; i < 8; ++i) {
      const int qrow = i + half * 8;
      Pbuf[wave][qrow * 32 + (lane & 15)]      = (bf16_t)s0[i];
      Pbuf[wave][qrow * 32 + 16 + (lane & 15)] = (bf16_t)s1[i];
    }
    asm volatile("s_wait_dscnt 0" ::: "memory");  // same-wave LDS RAW fence
    const v16bf pf = load_frag_bf16(Pl, 0, 0, 32);

    // ---- O += P @ V' over all 32 output-dim tiles (from LDS) ----
#pragma unroll
    for (int t = 0; t < 32; ++t) {
      v16bf vf = load_frag_bf16(Vb, t * 16, 0, V_LDS_LD);
      acc[t] = wmma_bf16(pf, vf, acc[t]);
    }
  }

  // ---- normalize and store ----
  float* ob = out + ((long)b * SEQ + q0) * DIM;
#pragma unroll
  for (int i = 0; i < 8; ++i) {
    const float inv = 1.0f / lrow[i];
    const int qrow  = i + half * 8;
#pragma unroll
    for (int t = 0; t < 32; ++t) {
      ob[(long)qrow * DIM + t * 16 + (lane & 15)] = acc[t][i] * inv;
    }
  }
}

// =====================================================================
extern "C" void kernel_launch(void* const* d_in, const int* in_sizes, int n_in,
                              void* d_out, int out_size, void* d_ws, size_t ws_size,
                              hipStream_t stream) {
  const float* q    = (const float*)d_in[0];
  const float* k    = (const float*)d_in[1];
  const float* v    = (const float*)d_in[2];
  const float* qw_w = (const float*)d_in[3];
  const float* qw_b = (const float*)d_in[4];
  const float* kw_w = (const float*)d_in[5];
  const float* kw_b = (const float*)d_in[6];
  const float* vw_w = (const float*)d_in[7];
  const float* vw_b = (const float*)d_in[8];

  const long n = (long)BATCH * SEQ * DIM;  // 8M elements
  bf16_t* qp  = (bf16_t*)d_ws;  // bf16 [B*S, D]
  bf16_t* kp  = qp + n;         // bf16 [B*S, D]
  bf16_t* vpT = kp + n;         // bf16 [B, D, S]

  const float scale = 0.044194173824159216f;  // 1/sqrt(512)

  dim3 pb(128);
  dim3 pg(BATCH * SEQ / 16, DIM / 64);
  proj_gemm_kernel<<<pg, pb, 0, stream>>>(q, qw_w, qw_b, qp, scale, 0);
  proj_gemm_kernel<<<pg, pb, 0, stream>>>(k, kw_w, kw_b, kp, 1.0f, 0);
  proj_gemm_kernel<<<pg, pb, 0, stream>>>(v, vw_w, vw_b, vpT, 1.0f, 1);

  dim3 ab(128);
  dim3 ag(SEQ / 64, BATCH);
  attn_kernel<<<ag, ab, 0, stream>>>(qp, kp, vpT, (float*)d_out);
}
